// EnergyTransformer_81011673137212
// MI455X (gfx1250) — matrix operand order repaired and checked by reading, not compiled, not run on hardware
//
#include <hip/hip_runtime.h>
#include <stdint.h>

typedef __bf16 bf16_t;
typedef __attribute__((ext_vector_type(16))) __bf16 v16bf;
typedef __attribute__((ext_vector_type(8)))  __bf16 v8bf;
typedef __attribute__((ext_vector_type(8)))  float  v8f;

#define D_MODEL 768
#define N_HEADS 12
#define D_HEAD  64
#define N_MEM   3072
#define SEQ     1024
#define BATCH   8
#define NTOK    (BATCH * SEQ)   /* 8192 tokens */
#define ALPHA   0.1f
#define EPS     1e-5f
#define NSTEPS  12

// ---------------------------------------------------------------------------
// WMMA helpers (CDNA5 wave32, v_wmma_f32_16x16x32_bf16)
// ---------------------------------------------------------------------------
__device__ __forceinline__ v8f vzero8() {
  v8f z;
#pragma unroll
  for (int i = 0; i < 8; ++i) z[i] = 0.f;
  return z;
}

__device__ __forceinline__ v8f wmma_bf16(v16bf a, v16bf b, v8f c) {
  return __builtin_amdgcn_wmma_f32_16x16x32_bf16(false, a, false, b,
                                                 (short)0, c, false, false);
}

// A operand: 16x32 (MxK) bf16 tile from row-major A (leading dim lda).
// ISA layout: lanes 0-15 -> kbase 0, lanes 16-31 -> kbase 8;
// VGPR v<4: K = kbase+2v(+pos);  v>=4: K = 16+kbase+2(v-4)(+pos).
// (Per lane this is two contiguous 16B chunks -> compiler emits 2x b128.)
__device__ __forceinline__ v16bf load_wmma_a(const bf16_t* __restrict__ A, int lda,
                                             int m0, int k0, int lane) {
  int m = lane & 15;
  int kbase = (lane & 16) ? 8 : 0;
  const bf16_t* row = A + (size_t)(m0 + m) * lda + k0;
  v16bf r;
#pragma unroll
  for (int v = 0; v < 8; ++v) {
    int k = (v < 4) ? (kbase + 2 * v) : (16 + kbase + 2 * (v - 4));
    r[2 * v]     = row[k];
    r[2 * v + 1] = row[k + 1];
  }
  return r;
}

// B operand: 32x16 (KxN) tile; Bt stored row-major as (N x K), leading dim ldb.
// Lane n holds column n; lanes 0-15 K=0..7,16..23, lanes 16-31 K=8..15,24..31.
__device__ __forceinline__ v16bf load_wmma_b(const bf16_t* __restrict__ Bt, int ldb,
                                             int n0, int k0, int lane) {
  int n = lane & 15;
  int koff = (lane & 16) ? 8 : 0;
  const bf16_t* p = Bt + (size_t)(n0 + n) * ldb + k0 + koff;
  const v8bf* p8 = (const v8bf*)p;            // 16B aligned by construction
  v8bf lo = p8[0];
  v8bf hi = p8[2];
  return __builtin_shufflevector(lo, hi, 0, 1, 2, 3, 4, 5, 6, 7,
                                 8, 9, 10, 11, 12, 13, 14, 15);
}

// Strided B operand: element (k=koff+(e<8?e:e+8), n) at base + n + k*estride.
__device__ __forceinline__ v16bf load_wmma_b_str(const bf16_t* __restrict__ base,
                                                 int estride, int lane) {
  int n = lane & 15;
  int koff = (lane & 16) ? 8 : 0;
  const bf16_t* p = base + n + (size_t)koff * estride;
  v16bf r;
#pragma unroll
  for (int e = 0; e < 16; ++e) r[e] = p[(size_t)(e < 8 ? e : e + 8) * estride];
  return r;
}

// ---------------------------------------------------------------------------
// Generic bf16 WMMA GEMM, 64x64 tile per wave (4x4 fragments, 16 WMMA per
// 32-deep K chunk; 4x register reuse on both operands).
// C[Mr x N] = scale * A[Mr x K] * Bt^T ; Bt given row-major as (N x K).
// mode: 0=store bf16, 1=relu->bf16, 2=store f32, 3=accumulate f32.
// Mr, N must be multiples of 64; K multiple of 32.
// ---------------------------------------------------------------------------
__global__ __launch_bounds__(256) void gemm_bf16_kernel(
    const bf16_t* __restrict__ A, int lda,
    const bf16_t* __restrict__ Bt, int ldb,
    void* __restrict__ Cp, int ldc,
    int Mr, int N, int K, int mode, float scale) {
  int lane = threadIdx.x & 31;
  int wave = threadIdx.x >> 5;
  int tile = blockIdx.x * 8 + wave;
  int tilesN = N >> 6;
  if (tile >= (Mr >> 6) * tilesN) return;
  int m0 = (tile / tilesN) << 6;
  int n0 = (tile % tilesN) << 6;

  v8f acc[4][4];
#pragma unroll
  for (int i = 0; i < 4; ++i)
#pragma unroll
    for (int j = 0; j < 4; ++j) acc[i][j] = vzero8();

  for (int k0 = 0; k0 < K; k0 += 32) {
    v16bf a[4], b[4];
#pragma unroll
    for (int i = 0; i < 4; ++i) a[i] = load_wmma_a(A, lda, m0 + 16 * i, k0, lane);
#pragma unroll
    for (int j = 0; j < 4; ++j) b[j] = load_wmma_b(Bt, ldb, n0 + 16 * j, k0, lane);
#pragma unroll
    for (int i = 0; i < 4; ++i)
#pragma unroll
      for (int j = 0; j < 4; ++j) acc[i][j] = wmma_bf16(a[i], b[j], acc[i][j]);
  }

  int n = lane & 15, hlf = lane >> 4;
  if (mode <= 1) {
    bf16_t* C = (bf16_t*)Cp;
#pragma unroll
    for (int i = 0; i < 4; ++i)
#pragma unroll
      for (int j = 0; j < 4; ++j)
#pragma unroll
        for (int v = 0; v < 8; ++v) {
          int m = v + 8 * hlf;
          float x = acc[i][j][v] * scale;
          if (mode == 1) x = fmaxf(x, 0.f);
          C[(size_t)(m0 + 16 * i + m) * ldc + n0 + 16 * j + n] = (bf16_t)x;
        }
  } else {
    float* C = (float*)Cp;
#pragma unroll
    for (int i = 0; i < 4; ++i)
#pragma unroll
      for (int j = 0; j < 4; ++j)
#pragma unroll
        for (int v = 0; v < 8; ++v) {
          int m = v + 8 * hlf;
          size_t idx = (size_t)(m0 + 16 * i + m) * ldc + n0 + 16 * j + n;
          float x = acc[i][j][v] * scale;
          if (mode == 3) x += C[idx];
          C[idx] = x;
        }
  }
}

// ---------------------------------------------------------------------------
// LayerNorm: per token compute y=(x-mean)*rstd, g = gamma*y + delta (bf16).
// ---------------------------------------------------------------------------
__global__ __launch_bounds__(256) void ln_kernel(
    const float* __restrict__ x, const float* __restrict__ gamma,
    const float* __restrict__ delta, float* __restrict__ y,
    float* __restrict__ rstd, bf16_t* __restrict__ gbf) {
  __shared__ float red[256];
  int t = blockIdx.x;
  int tid = threadIdx.x;
  const float* xr = x + (size_t)t * D_MODEL;

  float s = 0.f;
  for (int j = tid; j < D_MODEL; j += 256) s += xr[j];
  red[tid] = s; __syncthreads();
  for (int off = 128; off > 0; off >>= 1) {
    if (tid < off) red[tid] += red[tid + off];
    __syncthreads();
  }
  float mean = red[0] * (1.f / D_MODEL);
  __syncthreads();

  float s2 = 0.f;
  for (int j = tid; j < D_MODEL; j += 256) {
    float d = xr[j] - mean; s2 += d * d;
  }
  red[tid] = s2; __syncthreads();
  for (int off = 128; off > 0; off >>= 1) {
    if (tid < off) red[tid] += red[tid + off];
    __syncthreads();
  }
  float var = red[0] * (1.f / D_MODEL);
  float rs = rsqrtf(var + EPS);
  float g0 = gamma[0];
  for (int j = tid; j < D_MODEL; j += 256) {
    float yy = (xr[j] - mean) * rs;
    y[(size_t)t * D_MODEL + j] = yy;
    gbf[(size_t)t * D_MODEL + j] = (bf16_t)(g0 * yy + delta[j]);
  }
  if (tid == 0) rstd[t] = rs;
}

// ---------------------------------------------------------------------------
// Attention dQ: flash-style.  One wave per (b,h,16 q-rows).  Streams K tiles,
// online softmax (running m,l), dQ = sum_k softmax(beta*QK^T) * K.
// Writes dQ bf16 in token-major layout (B,S,H,Y) and final (m,l) stats.
// ---------------------------------------------------------------------------
__global__ __launch_bounds__(256) void attn_dq_kernel(
    const bf16_t* __restrict__ Qb, const bf16_t* __restrict__ Kb,
    const float* __restrict__ beta, bf16_t* __restrict__ dQ,
    float* __restrict__ mstat, float* __restrict__ lstat) {
  __shared__ float smem[8 * 640];
  int lane = threadIdx.x & 31;
  int wave = threadIdx.x >> 5;
  float* stile = smem + wave * 640;       // 16x32 scores
  float* fac   = stile + 512;             // [16] per-row rescale / 1/l
  float* mrowa = stile + 528;             // [16] per-row running max

  int tile = blockIdx.x * 8 + wave;       // 6144 tiles total (exact)
  int b  = tile / (N_HEADS * (SEQ / 16));
  int h  = (tile / (SEQ / 16)) % N_HEADS;
  int q0 = (tile % (SEQ / 16)) * 16;
  float betah = beta[h];
  int n = lane & 15, hlf = lane >> 4;

  const bf16_t* Qrow = Qb + (size_t)(b * SEQ + q0) * D_MODEL + h * D_HEAD;
  v16bf qa0 = load_wmma_a(Qrow, D_MODEL, 0, 0, lane);
  v16bf qa1 = load_wmma_a(Qrow, D_MODEL, 0, 32, lane);

  v8f acc[4];
#pragma unroll
  for (int t = 0; t < 4; ++t) acc[t] = vzero8();
  float mrun = -1e30f, lrun = 0.f;        // valid on lanes 0..15 (row = lane)

  for (int kc = 0; kc < SEQ; kc += 32) {
    // scores S[16q x 32k] = beta * Q K^T
#pragma unroll
    for (int kt = 0; kt < 32; kt += 16) {
      const bf16_t* Krow = Kb + (size_t)(b * SEQ + kc + kt) * D_MODEL + h * D_HEAD;
      v16bf kb0 = load_wmma_b(Krow, D_MODEL, 0, 0, lane);
      v16bf kb1 = load_wmma_b(Krow, D_MODEL, 0, 32, lane);
      v8f s = wmma_bf16(qa0, kb0, vzero8());
      s = wmma_bf16(qa1, kb1, s);
#pragma unroll
      for (int v = 0; v < 8; ++v)
        stile[(v + 8 * hlf) * 32 + kt + n] = s[v] * betah;
    }
    __syncthreads();

    if (lane < 16) {                      // per-row online softmax stats
      float mloc = -1e30f;
      for (int j = 0; j < 32; ++j) mloc = fmaxf(mloc, stile[lane * 32 + j]);
      float mnew = fmaxf(mrun, mloc);
      float f = __expf(mrun - mnew);
      float sum = 0.f;
      for (int j = 0; j < 32; ++j) sum += __expf(stile[lane * 32 + j] - mnew);
      lrun = lrun * f + sum;
      mrun = mnew;
      fac[lane] = f;
      mrowa[lane] = mnew;
    }
    __syncthreads();

    // rescale dQ accumulators
#pragma unroll
    for (int t = 0; t < 4; ++t)
#pragma unroll
      for (int v = 0; v < 8; ++v) acc[t][v] *= fac[v + 8 * hlf];

    // gather P = exp(s - m) into WMMA A-operand layout
    int m = lane & 15;
    int kbase = hlf * 8;
    float mr = mrowa[m];
    v16bf pa;
#pragma unroll
    for (int v = 0; v < 8; ++v) {
      int k = (v < 4) ? (kbase + 2 * v) : (16 + kbase + 2 * (v - 4));
      pa[2 * v]     = (bf16_t)__expf(stile[m * 32 + k] - mr);
      pa[2 * v + 1] = (bf16_t)__expf(stile[m * 32 + k + 1] - mr);
    }
    // dQ += P (16x32) * K (32 x 64)
#pragma unroll
    for (int nt = 0; nt < 4; ++nt) {
      const bf16_t* kbase_p =
          Kb + (size_t)(b * SEQ + kc) * D_MODEL + h * D_HEAD + nt * 16;
      v16bf kbv = load_wmma_b_str(kbase_p, D_MODEL, lane);
      acc[nt] = wmma_bf16(pa, kbv, acc[nt]);
    }
    __syncthreads();
  }

  if (lane < 16) fac[lane] = 1.f / lrun;
  __syncthreads();
#pragma unroll
  for (int t = 0; t < 4; ++t)
#pragma unroll
    for (int v = 0; v < 8; ++v) {
      int m = v + 8 * hlf;
      float x = acc[t][v] * fac[m];
      dQ[((size_t)(b * SEQ + q0 + m) * N_HEADS + h) * D_HEAD + t * 16 + n] =
          (bf16_t)x;
    }
  if (lane < 16) {
    int qi = (b * N_HEADS + h) * SEQ + q0 + lane;
    mstat[qi] = mrun;
    lstat[qi] = lrun;
  }
}

// ---------------------------------------------------------------------------
// Attention dK: one wave per (b,h,16 k-rows).  Streams Q tiles, uses stored
// (m,l) stats:  dK = sum_q P^T Q, P[q][k] = exp(beta*A - m_q)/l_q.
// ---------------------------------------------------------------------------
__global__ __launch_bounds__(256) void attn_dk_kernel(
    const bf16_t* __restrict__ Qb, const bf16_t* __restrict__ Kb,
    const float* __restrict__ beta, const float* __restrict__ mstat,
    const float* __restrict__ lstat, bf16_t* __restrict__ dK) {
  __shared__ float smem[8 * 640];
  int lane = threadIdx.x & 31;
  int wave = threadIdx.x >> 5;
  float* stile = smem + wave * 640;       // 16x32 scores (rows = k, cols = q)
  float* marr  = stile + 544;             // [32] per-q max
  float* larr  = stile + 576;             // [32] per-q sum

  int tile = blockIdx.x * 8 + wave;
  int b  = tile / (N_HEADS * (SEQ / 16));
  int h  = (tile / (SEQ / 16)) % N_HEADS;
  int k0 = (tile % (SEQ / 16)) * 16;
  float betah = beta[h];
  int n = lane & 15, hlf = lane >> 4;

  const bf16_t* Krow = Kb + (size_t)(b * SEQ + k0) * D_MODEL + h * D_HEAD;
  v16bf ka0 = load_wmma_a(Krow, D_MODEL, 0, 0, lane);
  v16bf ka1 = load_wmma_a(Krow, D_MODEL, 0, 32, lane);

  v8f acc[4];
#pragma unroll
  for (int t = 0; t < 4; ++t) acc[t] = vzero8();

  for (int qc = 0; qc < SEQ; qc += 32) {
    int qi = (b * N_HEADS + h) * SEQ + qc + lane;
    marr[lane] = mstat[qi];
    larr[lane] = lstat[qi];

    // T[16k x 32q] = beta * K Q^T
#pragma unroll
    for (int qt = 0; qt < 32; qt += 16) {
      const bf16_t* Qrow2 = Qb + (size_t)(b * SEQ + qc + qt) * D_MODEL + h * D_HEAD;
      v16bf qb0 = load_wmma_b(Qrow2, D_MODEL, 0, 0, lane);
      v16bf qb1 = load_wmma_b(Qrow2, D_MODEL, 0, 32, lane);
      v8f s = wmma_bf16(ka0, qb0, vzero8());
      s = wmma_bf16(ka1, qb1, s);
#pragma unroll
      for (int v = 0; v < 8; ++v)
        stile[(v + 8 * hlf) * 32 + qt + n] = s[v] * betah;
    }
    __syncthreads();

    // gather normalized P^T into A-operand layout
    int m = lane & 15;
    int kbase = hlf * 8;
    v16bf pa;
#pragma unroll
    for (int v = 0; v < 8; ++v) {
      int kk = (v < 4) ? (kbase + 2 * v) : (16 + kbase + 2 * (v - 4));
      float p0 = __expf(stile[m * 32 + kk] - marr[kk]) / larr[kk];
      float p1 = __expf(stile[m * 32 + kk + 1] - marr[kk + 1]) / larr[kk + 1];
      pa[2 * v]     = (bf16_t)p0;
      pa[2 * v + 1] = (bf16_t)p1;
    }
    // dK += P^T (16x32) * Q (32 x 64)
#pragma unroll
    for (int nt = 0; nt < 4; ++nt) {
      const bf16_t* qbase_p =
          Qb + (size_t)(b * SEQ + qc) * D_MODEL + h * D_HEAD + nt * 16;
      v16bf qbv = load_wmma_b_str(qbase_p, D_MODEL, lane);
      acc[nt] = wmma_bf16(pa, qbv, acc[nt]);
    }
    __syncthreads();
  }

#pragma unroll
  for (int t = 0; t < 4; ++t)
#pragma unroll
    for (int v = 0; v < 8; ++v) {
      int m = v + 8 * hlf;
      dK[((size_t)(b * SEQ + k0 + m) * N_HEADS + h) * D_HEAD + t * 16 + n] =
          (bf16_t)acc[t][v];
    }
}

// ---------------------------------------------------------------------------
// Jacobian-transpose step:  v = x + grad_g;  dx = gamma*rstd*(v - mean(v)
//                           - y*mean(y.v));  x -= ALPHA*dx.
// ---------------------------------------------------------------------------
__global__ __launch_bounds__(256) void jstep_kernel(
    float* __restrict__ xv, const float* __restrict__ grad,
    const float* __restrict__ y, const float* __restrict__ rstd,
    const float* __restrict__ gamma) {
  __shared__ float r1[256], r2[256];
  int t = blockIdx.x;
  int tid = threadIdx.x;
  float* xr = xv + (size_t)t * D_MODEL;
  const float* gr = grad + (size_t)t * D_MODEL;
  const float* yr = y + (size_t)t * D_MODEL;

  float s1 = 0.f, s2 = 0.f;
  for (int j = tid; j < D_MODEL; j += 256) {
    float v = xr[j] + gr[j];
    s1 += v;
    s2 += yr[j] * v;
  }
  r1[tid] = s1; r2[tid] = s2; __syncthreads();
  for (int off = 128; off > 0; off >>= 1) {
    if (tid < off) { r1[tid] += r1[tid + off]; r2[tid] += r2[tid + off]; }
    __syncthreads();
  }
  float mv  = r1[0] * (1.f / D_MODEL);
  float myv = r2[0] * (1.f / D_MODEL);
  float rs = rstd[t];
  float g0 = gamma[0];
  for (int j = tid; j < D_MODEL; j += 256) {
    float v = xr[j] + gr[j];
    float o = g0 * rs * (v - mv - yr[j] * myv);
    xr[j] = xr[j] - ALPHA * o;
  }
}

// ---------------------------------------------------------------------------
// One-shot f32 -> bf16 weight conversion (also builds transpose).
// ---------------------------------------------------------------------------
__global__ __launch_bounds__(256) void cvt_kernel(
    const float* __restrict__ src, bf16_t* __restrict__ dst,
    bf16_t* __restrict__ dstT, int rows, int cols) {
  int idx = blockIdx.x * 256 + threadIdx.x;
  if (idx >= rows * cols) return;
  int r = idx / cols, c = idx % cols;
  bf16_t v = (bf16_t)src[idx];
  dst[idx] = v;
  dstT[(size_t)c * rows + r] = v;
}

// ---------------------------------------------------------------------------
// Host driver
// ---------------------------------------------------------------------------
extern "C" void kernel_launch(void* const* d_in, const int* in_sizes, int n_in,
                              void* d_out, int out_size, void* d_ws, size_t ws_size,
                              hipStream_t stream) {
  (void)in_sizes; (void)n_in; (void)out_size; (void)ws_size;
  const float* x_in  = (const float*)d_in[0];
  const float* gamma = (const float*)d_in[1];
  const float* delta = (const float*)d_in[2];
  const float* Wq    = (const float*)d_in[3];
  const float* Wk    = (const float*)d_in[4];
  const float* beta  = (const float*)d_in[5];
  const float* xi    = (const float*)d_in[6];

  char* w = (char*)d_ws;
  auto alloc = [&](size_t bytes) -> char* {
    char* p = w;
    w += (bytes + 255) & ~(size_t)255;
    return p;
  };
  float*  ybuf   = (float*) alloc((size_t)NTOK * D_MODEL * 4);
  float*  rstdb  = (float*) alloc((size_t)NTOK * 4);
  bf16_t* gbf    = (bf16_t*)alloc((size_t)NTOK * D_MODEL * 2);
  bf16_t* Wq_bf  = (bf16_t*)alloc((size_t)D_MODEL * D_MODEL * 2);
  bf16_t* WqT_bf = (bf16_t*)alloc((size_t)D_MODEL * D_MODEL * 2);
  bf16_t* Wk_bf  = (bf16_t*)alloc((size_t)D_MODEL * D_MODEL * 2);
  bf16_t* WkT_bf = (bf16_t*)alloc((size_t)D_MODEL * D_MODEL * 2);
  bf16_t* xi_bf  = (bf16_t*)alloc((size_t)N_MEM * D_MODEL * 2);
  bf16_t* xiT_bf = (bf16_t*)alloc((size_t)N_MEM * D_MODEL * 2);
  bf16_t* Qbf    = (bf16_t*)alloc((size_t)NTOK * D_MODEL * 2);
  bf16_t* Kbf    = (bf16_t*)alloc((size_t)NTOK * D_MODEL * 2);
  bf16_t* dQbf   = (bf16_t*)alloc((size_t)NTOK * D_MODEL * 2);
  bf16_t* dKbf   = (bf16_t*)alloc((size_t)NTOK * D_MODEL * 2);
  float*  mstat  = (float*) alloc((size_t)BATCH * N_HEADS * SEQ * 4);
  float*  lstat  = (float*) alloc((size_t)BATCH * N_HEADS * SEQ * 4);
  float*  gradb  = (float*) alloc((size_t)NTOK * D_MODEL * 4);
  bf16_t* hbf    = (bf16_t*)alloc((size_t)NTOK * N_MEM * 2);
  float*  xbuf   = (float*) alloc((size_t)NTOK * D_MODEL * 4);

  auto gemm = [&](const bf16_t* A, int lda, const bf16_t* Bt, int ldb, void* C,
                  int ldc, int Mr, int N, int K, int mode, float scale) {
    int tiles = (Mr / 64) * (N / 64);
    int blocks = (tiles + 7) / 8;
    gemm_bf16_kernel<<<blocks, 256, 0, stream>>>(A, lda, Bt, ldb, C, ldc,
                                                 Mr, N, K, mode, scale);
  };

  // setup: weights to bf16 (+ transposes), state init
  int we = D_MODEL * D_MODEL;
  cvt_kernel<<<(we + 255) / 256, 256, 0, stream>>>(Wq, Wq_bf, WqT_bf, D_MODEL, D_MODEL);
  cvt_kernel<<<(we + 255) / 256, 256, 0, stream>>>(Wk, Wk_bf, WkT_bf, D_MODEL, D_MODEL);
  int xe = N_MEM * D_MODEL;
  cvt_kernel<<<(xe + 255) / 256, 256, 0, stream>>>(xi, xi_bf, xiT_bf, N_MEM, D_MODEL);
  (void)hipMemcpyAsync(xbuf, x_in, (size_t)NTOK * D_MODEL * 4,
                       hipMemcpyDeviceToDevice, stream);

  int attn_blocks = (BATCH * N_HEADS * (SEQ / 16)) / 8;   // 768, exact

  for (int step = 0; step < NSTEPS; ++step) {
    ln_kernel<<<NTOK, 256, 0, stream>>>(xbuf, gamma, delta, ybuf, rstdb, gbf);

    // Q/K projections: (8192x768) x (768x768)
    gemm(gbf, D_MODEL, Wq_bf, D_MODEL, Qbf, D_MODEL, NTOK, D_MODEL, D_MODEL, 0, 1.f);
    gemm(gbf, D_MODEL, Wk_bf, D_MODEL, Kbf, D_MODEL, NTOK, D_MODEL, D_MODEL, 0, 1.f);

    attn_dq_kernel<<<attn_blocks, 256, 0, stream>>>(Qbf, Kbf, beta, dQbf, mstat, lstat);
    attn_dk_kernel<<<attn_blocks, 256, 0, stream>>>(Qbf, Kbf, beta, mstat, lstat, dKbf);

    // grad_g = -(dQ Wq + dK Wk)
    gemm(dQbf, D_MODEL, WqT_bf, D_MODEL, gradb, D_MODEL, NTOK, D_MODEL, D_MODEL, 2, -1.f);
    gemm(dKbf, D_MODEL, WkT_bf, D_MODEL, gradb, D_MODEL, NTOK, D_MODEL, D_MODEL, 3, -1.f);

    // Hopfield: h = relu(g xi^T);  grad_g -= h xi
    gemm(gbf, D_MODEL, xi_bf, D_MODEL, hbf, N_MEM, NTOK, N_MEM, D_MODEL, 1, 1.f);
    gemm(hbf, N_MEM, xiT_bf, N_MEM, gradb, D_MODEL, NTOK, D_MODEL, N_MEM, 3, -1.f);

    jstep_kernel<<<NTOK, 256, 0, stream>>>(xbuf, gradb, ybuf, rstdb, gamma);
  }

  (void)hipMemcpyAsync(d_out, xbuf, (size_t)NTOK * D_MODEL * 4,
                       hipMemcpyDeviceToDevice, stream);
}
